// FreqEnocder_11038065951111
// MI455X (gfx1250) — matrix operand order repaired and checked by reading, not compiled, not run on hardware
//
#include <hip/hip_runtime.h>
#include <hip/hip_bf16.h>

typedef __attribute__((ext_vector_type(16))) _Float16 v16h;
typedef __attribute__((ext_vector_type(8)))  _Float16 v8h;
typedef __attribute__((ext_vector_type(8)))  float    v8f;
typedef __attribute__((ext_vector_type(4)))  int      v4i;

#define BATCH 512
#define P1    64
#define LDIM  512
#define HDIM  2048
#define ROWS  (BATCH * P1)          // 32768
#define SLAB  (P1 * LDIM)           // 32768 elems per LayerNorm slab
#define INV_SQRT_L 0.04419417382415922f  // 1/sqrt(512)

#define LDS_STRIDE 40               // padded row stride (halves) for bank spread
#define LDS_BUF    (64 * LDS_STRIDE)

#if __has_builtin(__builtin_amdgcn_global_load_async_to_lds_b128)
#define FE_ASYNC_LDS 1
#else
#define FE_ASYNC_LDS 0
#endif

// 16-byte memory -> LDS stage; async path bypasses VGPRs (ASYNCcnt tracked)
static __device__ __forceinline__ void fe_stage16(const _Float16* g, _Float16* l) {
#if FE_ASYNC_LDS
    __builtin_amdgcn_global_load_async_to_lds_b128((v4i*)g, (v4i*)l, 0, 0);
#else
    *reinterpret_cast<v8h*>(l) = *reinterpret_cast<const v8h*>(g);
#endif
}

static __device__ __forceinline__ void fe_wait_stage() {
#if FE_ASYNC_LDS
#if __has_builtin(__builtin_amdgcn_s_wait_asynccnt)
    __builtin_amdgcn_s_wait_asynccnt(0);
#else
    asm volatile("s_wait_asynccnt 0x0" ::: "memory");
#endif
#endif
}

// ---------------------------------------------------------------------------
// Fragment loaders (per CDNA5 ISA 7.12.2 VGPR layouts, wave32)
// ---------------------------------------------------------------------------
static __device__ __forceinline__ v16h load_a_frag(const _Float16* A, int lda,
                                                   int row0, int k0, int lane) {
    int m = lane & 15, sel = lane >> 4;
    const _Float16* p = A + (size_t)(row0 + m) * lda + k0 + 8 * sel;
    v8h lo = *reinterpret_cast<const v8h*>(p);
    v8h hi = *reinterpret_cast<const v8h*>(p + 16);
    return __builtin_shufflevector(lo, hi, 0,1,2,3,4,5,6,7,8,9,10,11,12,13,14,15);
}

// B fragment from LDS stage: lb holds 64 cols x 32 k (padded rows)
static __device__ __forceinline__ v16h load_b_lds(const _Float16* lb, int t, int lane) {
    int n = lane & 15, sel = lane >> 4;
    const _Float16* p = lb + (size_t)(t * 16 + n) * LDS_STRIDE + 8 * sel;
    v8h lo = *reinterpret_cast<const v8h*>(p);
    v8h hi = *reinterpret_cast<const v8h*>(p + 16);
    return __builtin_shufflevector(lo, hi, 0,1,2,3,4,5,6,7,8,9,10,11,12,13,14,15);
}

#define FE_WMMA(a, b, c) __builtin_amdgcn_wmma_f32_16x16x32_f16( \
    false, (a), false, (b), (short)0, (c), false, false)

// ---------------------------------------------------------------------------
// GEMM core: block = 256 threads = 8 waves; block tile 256 rows x 64 cols.
// Wave tile 32x64 (2 A frags x 4 B frags -> 8 accumulators).
// B (N-major, ldb=KD) staged via double-buffered async global->LDS copies
// issued one K-step ahead; A frags prefetched one K-step ahead in registers.
// ---------------------------------------------------------------------------
template<int KD, int LDA>
static __device__ __forceinline__ void gemm_core(const _Float16* __restrict__ A,
                                                 const _Float16* __restrict__ Bt,
                                                 _Float16* lds,   // [2][LDS_BUF]
                                                 int row0, int n0, int tid,
                                                 v8f acc[2][4]) {
    int lane = tid & 31;
    // cooperative staging coords: thread -> (col 0..63, 16B chunk 0..3)
    int sc  = tid >> 2;
    int sch = tid & 3;
    const _Float16* sp = Bt + (size_t)(n0 + sc) * KD + sch * 8;
    _Float16* sd0 = lds + (size_t)sc * LDS_STRIDE + sch * 8;

    // prologue: stage k=0 into buffer 0
    fe_stage16(sp, sd0);
    fe_wait_stage();
    __syncthreads();

    v16h a0 = load_a_frag(A, LDA, row0,      0, lane);
    v16h a1 = load_a_frag(A, LDA, row0 + 16, 0, lane);
    int buf = 0;
    for (int k0 = 0; k0 < KD; k0 += 32) {
        const bool more = (k0 + 32) < KD;
        v16h a0n = a0, a1n = a1;
        if (more) {
            // stage next K-step into the other buffer (async, no VGPR traffic)
            fe_stage16(sp + k0 + 32, sd0 + (size_t)(buf ^ 1) * LDS_BUF);
            a0n = load_a_frag(A, LDA, row0,      k0 + 32, lane);
            a1n = load_a_frag(A, LDA, row0 + 16, k0 + 32, lane);
        }
        const _Float16* lb = lds + (size_t)buf * LDS_BUF;
        v16h b0 = load_b_lds(lb, 0, lane);
        v16h b1 = load_b_lds(lb, 1, lane);
        v16h b2 = load_b_lds(lb, 2, lane);
        v16h b3 = load_b_lds(lb, 3, lane);
        acc[0][0] = FE_WMMA(a0, b0, acc[0][0]);
        acc[1][0] = FE_WMMA(a1, b0, acc[1][0]);
        acc[0][1] = FE_WMMA(a0, b1, acc[0][1]);
        acc[1][1] = FE_WMMA(a1, b1, acc[1][1]);
        acc[0][2] = FE_WMMA(a0, b2, acc[0][2]);
        acc[1][2] = FE_WMMA(a1, b2, acc[1][2]);
        acc[0][3] = FE_WMMA(a0, b3, acc[0][3]);
        acc[1][3] = FE_WMMA(a1, b3, acc[1][3]);
        fe_wait_stage();        // staged copy landed in LDS
        __syncthreads();        // all waves done reading buf, stage visible
        a0 = a0n; a1 = a1n; buf ^= 1;
    }
}

// ---------------------------------------------------------------------------
// LayerNorm over a [P1, LDIM] slab per block; writes f16 normalized output.
// ---------------------------------------------------------------------------
__global__ __launch_bounds__(256)
void fe_ln2d_kernel(const float* __restrict__ x, const float* __restrict__ g,
                    const float* __restrict__ beta, _Float16* __restrict__ out) {
    __shared__ float r1[256];
    __shared__ float r2[256];
    __shared__ float mu_s, rs_s;
    int b = blockIdx.x, tid = threadIdx.x;
    const float* xb = x + (size_t)b * SLAB;
    float s = 0.f, s2 = 0.f;
    for (int i = tid; i < SLAB; i += 256) {
        float v = xb[i];
        s += v; s2 += v * v;
    }
    r1[tid] = s; r2[tid] = s2;
    __syncthreads();
    for (int off = 128; off > 0; off >>= 1) {
        if (tid < off) { r1[tid] += r1[tid + off]; r2[tid] += r2[tid + off]; }
        __syncthreads();
    }
    if (tid == 0) {
        float mu = r1[0] * (1.f / SLAB);
        float var = r2[0] * (1.f / SLAB) - mu * mu;
        mu_s = mu;
        rs_s = rsqrtf(var + 1e-5f);
    }
    __syncthreads();
    float mu = mu_s, rs = rs_s;
    _Float16* ob = out + (size_t)b * SLAB;
    for (int i = tid; i < SLAB; i += 256)
        ob[i] = (_Float16)((xb[i] - mu) * rs * g[i] + beta[i]);
}

// ---------------------------------------------------------------------------
// Build transposed circulant kernel: Kt[n][k] = w[(n-k) mod 512] / sqrt(512)
// ---------------------------------------------------------------------------
__global__ __launch_bounds__(256)
void fe_build_kt_kernel(const float* __restrict__ w, _Float16* __restrict__ kt) {
    int n = blockIdx.x;
    for (int k = threadIdx.x; k < LDIM; k += 256)
        kt[(size_t)n * LDIM + k] = (_Float16)(w[(n - k) & (LDIM - 1)] * INV_SQRT_L);
}

// ---------------------------------------------------------------------------
// Transpose + f32->f16 convert: src [R, C] row-major -> dst [C, R]
// ---------------------------------------------------------------------------
__global__ __launch_bounds__(256)
void fe_transpose_f16_kernel(const float* __restrict__ src, _Float16* __restrict__ dst,
                             int R, int C) {
    int n = blockIdx.x;
    for (int k = threadIdx.x; k < R; k += 256)
        dst[(size_t)n * R + k] = (_Float16)(src[(size_t)k * C + n]);
}

// ---------------------------------------------------------------------------
// GEMM 1: circulant conv.  C(f32)[ROWS,512] = A(f16)[ROWS,512] x Kt^T
// ---------------------------------------------------------------------------
__global__ __launch_bounds__(256)
void fe_gemm_conv_kernel(const _Float16* __restrict__ A, const _Float16* __restrict__ Bt,
                         float* __restrict__ C) {
    __shared__ _Float16 lds[2 * LDS_BUF];
    int tid = threadIdx.x, lane = tid & 31, wv = tid >> 5;
    int row0 = blockIdx.x * 256 + wv * 32;
    int n0   = blockIdx.y * 64;
    v8f acc[2][4] = {};
    gemm_core<LDIM, LDIM>(A, Bt, lds, row0, n0, tid, acc);
    int n = lane & 15, sel = lane >> 4;
#pragma unroll
    for (int i = 0; i < 2; ++i)
#pragma unroll
        for (int t = 0; t < 4; ++t)
#pragma unroll
            for (int r = 0; r < 8; ++r)
                C[(size_t)(row0 + i * 16 + r + 8 * sel) * LDIM + n0 + 16 * t + n] =
                    acc[i][t][r];
}

// ---------------------------------------------------------------------------
// GEMM 2: FFN layer 1.  act(f16)[ROWS,2048] = relu(A[ROWS,512] x w1 + b1)
// ---------------------------------------------------------------------------
__global__ __launch_bounds__(256)
void fe_gemm_ffn1_kernel(const _Float16* __restrict__ A, const _Float16* __restrict__ Bt,
                         const float* __restrict__ bias, _Float16* __restrict__ act) {
    __shared__ _Float16 lds[2 * LDS_BUF];
    int tid = threadIdx.x, lane = tid & 31, wv = tid >> 5;
    int row0 = blockIdx.x * 256 + wv * 32;
    int n0   = blockIdx.y * 64;
    v8f acc[2][4] = {};
    gemm_core<LDIM, LDIM>(A, Bt, lds, row0, n0, tid, acc);
    int n = lane & 15, sel = lane >> 4;
#pragma unroll
    for (int t = 0; t < 4; ++t) {
        int col = n0 + 16 * t + n;
        float bb = bias[col];
#pragma unroll
        for (int i = 0; i < 2; ++i)
#pragma unroll
            for (int r = 0; r < 8; ++r) {
                float v = acc[i][t][r] + bb;
                v = v > 0.f ? v : 0.f;
                act[(size_t)(row0 + i * 16 + r + 8 * sel) * HDIM + col] = (_Float16)v;
            }
    }
}

// ---------------------------------------------------------------------------
// GEMM 3: FFN layer 2 + bias + residual.  out(f32)[ROWS,512]
// ---------------------------------------------------------------------------
__global__ __launch_bounds__(256)
void fe_gemm_ffn2_kernel(const _Float16* __restrict__ A, const _Float16* __restrict__ Bt,
                         const float* __restrict__ bias, const float* __restrict__ resid,
                         float* __restrict__ out) {
    __shared__ _Float16 lds[2 * LDS_BUF];
    int tid = threadIdx.x, lane = tid & 31, wv = tid >> 5;
    int row0 = blockIdx.x * 256 + wv * 32;
    int n0   = blockIdx.y * 64;
    v8f acc[2][4] = {};
    gemm_core<HDIM, HDIM>(A, Bt, lds, row0, n0, tid, acc);
    int n = lane & 15, sel = lane >> 4;
#pragma unroll
    for (int t = 0; t < 4; ++t) {
        int col = n0 + 16 * t + n;
        float bb = bias[col];
#pragma unroll
        for (int i = 0; i < 2; ++i)
#pragma unroll
            for (int r = 0; r < 8; ++r) {
                size_t idx = (size_t)(row0 + i * 16 + r + 8 * sel) * LDIM + col;
                out[idx] = acc[i][t][r] + bb + resid[idx];
            }
    }
}

// ---------------------------------------------------------------------------
// Host launcher
// ---------------------------------------------------------------------------
extern "C" void kernel_launch(void* const* d_in, const int* in_sizes, int n_in,
                              void* d_out, int out_size, void* d_ws, size_t ws_size,
                              hipStream_t stream) {
    const float* x     = (const float*)d_in[0];
    const float* w     = (const float*)d_in[1];
    const float* ln1_w = (const float*)d_in[2];
    const float* ln1_b = (const float*)d_in[3];
    const float* ln2_w = (const float*)d_in[4];
    const float* ln2_b = (const float*)d_in[5];
    const float* w1    = (const float*)d_in[6];
    const float* b1    = (const float*)d_in[7];
    const float* w2    = (const float*)d_in[8];
    const float* b2    = (const float*)d_in[9];
    float* out = (float*)d_out;

    // workspace layout
    char* ws = (char*)d_ws;
    size_t off = 0;
    _Float16* h1  = (_Float16*)(ws + off); off += (size_t)ROWS * LDIM * 2;   // 32 MB
    _Float16* kt  = (_Float16*)(ws + off); off += (size_t)LDIM * LDIM * 2;   // 0.5 MB
    float*    h2  = (float*)   (ws + off); off += (size_t)ROWS * LDIM * 4;   // 64 MB
    _Float16* h3  = (_Float16*)(ws + off); off += (size_t)ROWS * LDIM * 2;   // 32 MB
    _Float16* act = (_Float16*)(ws + off); off += (size_t)ROWS * HDIM * 2;   // 128 MB
    _Float16* w1t = (_Float16*)(ws + off); off += (size_t)HDIM * LDIM * 2;   // 2 MB
    _Float16* w2t = (_Float16*)(ws + off); off += (size_t)LDIM * HDIM * 2;   // 2 MB
    (void)ws_size; (void)in_sizes; (void)n_in; (void)out_size;

    // Stage 0: weight prep (independent)
    fe_build_kt_kernel<<<LDIM, 256, 0, stream>>>(w, kt);
    fe_transpose_f16_kernel<<<HDIM, 256, 0, stream>>>(w1, w1t, LDIM, HDIM);
    fe_transpose_f16_kernel<<<LDIM, 256, 0, stream>>>(w2, w2t, HDIM, LDIM);

    // Stage 1: LayerNorm1 (f32 -> f16)
    fe_ln2d_kernel<<<BATCH, 256, 0, stream>>>(x, ln1_w, ln1_b, h1);

    // Stage 2: circulant-conv GEMM  h2 = h1 @ K
    fe_gemm_conv_kernel<<<dim3(ROWS / 256, LDIM / 64), 256, 0, stream>>>(h1, kt, h2);

    // Stage 3: LayerNorm2 (f32 -> f16)
    fe_ln2d_kernel<<<BATCH, 256, 0, stream>>>(h2, ln2_w, ln2_b, h3);

    // Stage 4: FFN layer 1 + bias + ReLU
    fe_gemm_ffn1_kernel<<<dim3(ROWS / 256, HDIM / 64), 256, 0, stream>>>(h3, w1t, b1, act);

    // Stage 5: FFN layer 2 + bias + residual
    fe_gemm_ffn2_kernel<<<dim3(ROWS / 256, LDIM / 64), 256, 0, stream>>>(act, w2t, b2, x, out);
}
